// MultiHeadAttention_6081673691138
// MI455X (gfx1250) — matrix operand order repaired and checked by reading, not compiled, not run on hardware
//
#include <hip/hip_runtime.h>

// Problem constants (match reference).
#define Bb  4
#define Ss  1024
#define DMm 1024
#define Hh  16
#define DKk 64
#define DVv 64

typedef __attribute__((ext_vector_type(16))) _Float16 v16h;
typedef __attribute__((ext_vector_type(8)))  float    v8f;

#define WMMA_F16(a, b, c) \
  __builtin_amdgcn_wmma_f32_16x16x32_f16(false, (a), false, (b), (short)0, (c), false, false)

static __device__ __forceinline__ v8f v8f_zero() {
  v8f z;
#pragma unroll
  for (int i = 0; i < 8; ++i) z[i] = 0.0f;
  return z;
}

// A-matrix fragment, 16x32 f16, rows m0..m0+15, cols k0..k0+31 of row-major A (ld elems).
// ISA layout: lanes 0-15 M=lane, lanes 16-31 same M; VGPR j<4: K=2j+8*half, j>=4: K=16+2(j-4)+8*half.
// => per lane: two contiguous 8-halfword (16B) runs at k0+8*half and k0+16+8*half.
static __device__ __forceinline__ v16h load_a_frag(const _Float16* A, int ld, int m0, int k0) {
  const int lane = threadIdx.x & 31;
  const int m    = m0 + (lane & 15);
  const int half = lane >> 4;
  const _Float16* p = A + (size_t)m * ld + k0 + 8 * half;
  v16h a;
#pragma unroll
  for (int i = 0; i < 8; ++i) { a[i] = p[i]; a[8 + i] = p[16 + i]; }
  return a;
}

// B-matrix fragment, 32x16 f16, from a TRANSPOSED operand BT[n][k] (ld elems):
// lane n = n0+(lane&15); k = k0 + 8*half + (0..7) and k0 + 16 + 8*half + (0..7) (matches A-frag order).
static __device__ __forceinline__ v16h load_b_frag(const _Float16* BT, int ld, int n0, int k0) {
  const int lane = threadIdx.x & 31;
  const int n    = n0 + (lane & 15);
  const int half = lane >> 4;
  const _Float16* p = BT + (size_t)n * ld + k0 + 8 * half;
  v16h b;
#pragma unroll
  for (int i = 0; i < 8; ++i) { b[i] = p[i]; b[8 + i] = p[16 + i]; }
  return b;
}

// ---------------------------------------------------------------------------
// Kernel 1: fp32 -> f16 conversions + weight transposes into WMMA-friendly BT
// layouts. wT = [3][H][DK][DM] (k contiguous), woT = [DM][H*DV] (k contiguous).
// ---------------------------------------------------------------------------
__global__ void convert_kernel(const float* __restrict__ x,
                               const float* __restrict__ Wq,
                               const float* __restrict__ Wk,
                               const float* __restrict__ Wv,
                               const float* __restrict__ Wo,
                               _Float16* __restrict__ xh,
                               _Float16* __restrict__ wT,
                               _Float16* __restrict__ woT) {
  const size_t tid    = (size_t)blockIdx.x * blockDim.x + threadIdx.x;
  const size_t stride = (size_t)gridDim.x * blockDim.x;

  const size_t NX = (size_t)Bb * Ss * DMm;
  for (size_t i = tid; i < NX; i += stride) xh[i] = (_Float16)x[i];

  const size_t NW = (size_t)Hh * DMm * DKk;  // one of Wq/Wk/Wv
  for (size_t i = tid; i < 3 * NW; i += stride) {
    const int which = (int)(i / NW);
    const size_t r  = i % NW;
    const int k = (int)(r % DMm);
    const int n = (int)((r / DMm) % DKk);
    const int h = (int)(r / ((size_t)DMm * DKk));
    const float* src = (which == 0) ? Wq : (which == 1) ? Wk : Wv;
    wT[i] = (_Float16)src[((size_t)h * DMm + k) * DKk + n];
  }

  const size_t NO = (size_t)DMm * DMm;  // Wo is [H*DV=1024][DM=1024]
  for (size_t i = tid; i < NO; i += stride) {
    const int k = (int)(i % DMm);
    const int n = (int)(i / DMm);
    woT[i] = (_Float16)Wo[(size_t)k * DMm + n];
  }
}

// ---------------------------------------------------------------------------
// Kernel 2: QKV projections. grid = (B*S/128, H, 3). Each wave computes a
// 32-row x 64-col stripe: per k-step 2 A-frags + 4 B-frags feed 8 WMMAs
// (12 b128 loads / 8 wmma). B-frags live in distinct registers so loads
// overlap WMMA execution instead of forcing s_wait_loadcnt 0.
// Q,K stored row-major [B,H,S,DK]; V stored transposed [B,H,DV,S].
// ---------------------------------------------------------------------------
__global__ void qkv_kernel(const _Float16* __restrict__ xh,
                           const _Float16* __restrict__ wT,
                           const float* __restrict__ bq,
                           const float* __restrict__ bk,
                           const float* __restrict__ bv,
                           _Float16* __restrict__ qh,
                           _Float16* __restrict__ kh,
                           _Float16* __restrict__ vT) {
  const int wave  = threadIdx.x >> 5;
  const int lane  = threadIdx.x & 31;
  const int h     = blockIdx.y;
  const int which = blockIdx.z;
  const int row0  = blockIdx.x * 128 + wave * 32;  // global row in [0, B*S)
  const _Float16* wbase = wT + ((size_t)(which * Hh + h)) * DKk * DMm;

  v8f acc[2][4];
#pragma unroll
  for (int rb = 0; rb < 2; ++rb)
#pragma unroll
    for (int t = 0; t < 4; ++t) acc[rb][t] = v8f_zero();

  for (int kt = 0; kt < DMm; kt += 32) {
    if (kt + 32 < DMm) {
      __builtin_prefetch(xh + (size_t)(row0 + (lane & 15)) * DMm + kt + 32, 0, 3);
      __builtin_prefetch(xh + (size_t)(row0 + 16 + (lane & 15)) * DMm + kt + 32, 0, 3);
    }
    v16h a0 = load_a_frag(xh, DMm, row0, kt);
    v16h a1 = load_a_frag(xh, DMm, row0 + 16, kt);
    v16h bf[4];
#pragma unroll
    for (int t = 0; t < 4; ++t) bf[t] = load_b_frag(wbase, DMm, t * 16, kt);
#pragma unroll
    for (int t = 0; t < 4; ++t) {
      acc[0][t] = WMMA_F16(a0, bf[t], acc[0][t]);
      acc[1][t] = WMMA_F16(a1, bf[t], acc[1][t]);
    }
  }

  const float* bias = (which == 0) ? bq : (which == 1) ? bk : bv;
  const int half = lane >> 4;
  const int nloc = lane & 15;
#pragma unroll
  for (int rb = 0; rb < 2; ++rb) {
#pragma unroll
    for (int t = 0; t < 4; ++t) {
      const int n = t * 16 + nloc;
      const float bsc = bias[h * DKk + n];
#pragma unroll
      for (int r = 0; r < 8; ++r) {
        const int row = row0 + rb * 16 + r + 8 * half;
        const int b = row / Ss, s = row % Ss;
        const float val = acc[rb][t][r] + bsc;
        const size_t bh = (size_t)(b * Hh + h);
        if (which == 0)      qh[(bh * Ss + s) * DKk + n] = (_Float16)val;
        else if (which == 1) kh[(bh * Ss + s) * DKk + n] = (_Float16)val;
        else                 vT[(bh * DVv + n) * Ss + s] = (_Float16)val;
      }
    }
  }
}

// ---------------------------------------------------------------------------
// Kernel 3: attention. grid = (S/64, H, B), 4 waves; each wave owns 16 query
// rows end-to-end (scores -> LDS, softmax, wt fp32 out, wt x V) so no block
// barriers are needed. Dynamic LDS: 64 x 1024 f16 = 128 KB (2 WG / WGP).
// ---------------------------------------------------------------------------
__global__ void attn_kernel(const _Float16* __restrict__ qh,
                            const _Float16* __restrict__ kh,
                            const _Float16* __restrict__ vT,
                            const unsigned char* __restrict__ mk,
                            float* __restrict__ wt,
                            _Float16* __restrict__ oh) {
  extern __shared__ _Float16 smem[];  // [64][Ss]
  const int wave = threadIdx.x >> 5;
  const int lane = threadIdx.x & 31;
  const int b = blockIdx.z, h = blockIdx.y;
  const int m0 = blockIdx.x * 64;
  const size_t bh = (size_t)(b * Hh + h);
  const _Float16* qbh = qh + bh * Ss * DKk;
  const _Float16* kbh = kh + bh * Ss * DKk;
  const _Float16* vbh = vT + bh * DVv * Ss;
  float* wtbh = wt + bh * (size_t)Ss * Ss;

  const int mrow = m0 + wave * 16;
  const int half = lane >> 4;
  const int nloc = lane & 15;
  _Float16* lrow = smem + (size_t)wave * 16 * Ss;  // this wave's 16-row slice

  // Q fragments reused across all 64 key tiles (DK=64 -> two 32-wide k-steps).
  const v16h aq0 = load_a_frag(qbh, DKk, mrow, 0);
  const v16h aq1 = load_a_frag(qbh, DKk, mrow, 32);

  // Phase A: scores = Q.K^T / sqrt(DK), masked, into LDS (f16).
  for (int nt = 0; nt < Ss / 16; ++nt) {
    const int n0 = nt * 16;
    v16h b0 = load_b_frag(kbh, DKk, n0, 0);
    v16h b1 = load_b_frag(kbh, DKk, n0, 32);
    v8f acc = v8f_zero();
    acc = WMMA_F16(aq0, b0, acc);
    acc = WMMA_F16(aq1, b1, acc);
    const int t = n0 + nloc;
    const bool keep = mk[b * Ss + t] != 0;
#pragma unroll
    for (int r = 0; r < 8; ++r) {
      float val = acc[r] * 0.125f;           // 1/sqrt(64)
      if (!keep) val = -30000.0f;            // f16-representable "-inf"
      lrow[(size_t)(r + 8 * half) * Ss + t] = (_Float16)val;
    }
  }

  // Phase B: per-row softmax (wave-local; rows are private to this wave).
  for (int r = 0; r < 16; ++r) {
    _Float16* rowp = lrow + (size_t)r * Ss;
    float mx = -3.0e30f;
    for (int t = lane; t < Ss; t += 32) mx = fmaxf(mx, (float)rowp[t]);
#pragma unroll
    for (int off = 16; off > 0; off >>= 1) mx = fmaxf(mx, __shfl_xor(mx, off));
    float sum = 0.0f;
    for (int t = lane; t < Ss; t += 32) {
      const float e = __expf((float)rowp[t] - mx);
      rowp[t] = (_Float16)e;
      sum += e;
    }
#pragma unroll
    for (int off = 16; off > 0; off >>= 1) sum += __shfl_xor(sum, off);
    const float inv = 1.0f / sum;
    float* wrow = wtbh + (size_t)(mrow + r) * Ss;
    for (int t = lane; t < Ss; t += 32) {
      const float w = (float)rowp[t] * inv;
      rowp[t] = (_Float16)w;                 // normalized weights for wt x V
      wrow[t] = w;                           // fp32 wt output
    }
  }

  // Phase C: O = wt x V. A-frags straight from LDS, B-frags from vT [DV][S];
  // all 4 B-frags loaded before the WMMA burst so loads overlap matrix ops.
  v8f oacc[4];
#pragma unroll
  for (int t = 0; t < 4; ++t) oacc[t] = v8f_zero();
  for (int kt = 0; kt < Ss; kt += 32) {
    v16h a;
    const _Float16* p = lrow + (size_t)(lane & 15) * Ss + kt + 8 * half;
#pragma unroll
    for (int i = 0; i < 8; ++i) { a[i] = p[i]; a[8 + i] = p[16 + i]; }
    v16h bf[4];
#pragma unroll
    for (int t = 0; t < 4; ++t) bf[t] = load_b_frag(vbh, Ss, t * 16, kt);
#pragma unroll
    for (int t = 0; t < 4; ++t) oacc[t] = WMMA_F16(a, bf[t], oacc[t]);
  }
#pragma unroll
  for (int t = 0; t < 4; ++t) {
    const int n = t * 16 + nloc;
#pragma unroll
    for (int r = 0; r < 8; ++r) {
      const int s = mrow + r + 8 * half;
      oh[((size_t)b * Ss + s) * (Hh * DVv) + h * DVv + n] = (_Float16)oacc[t][r];
    }
  }
}

// ---------------------------------------------------------------------------
// Kernel 4: output projection out = oh[B*S,1024] x Wo + bo (fp32 out).
// grid = (B*S/128, DM/64); 32-row x 64-col stripe per wave, same pipelined
// inner loop shape as qkv (12 b128 loads / 8 wmma per k-step).
// ---------------------------------------------------------------------------
__global__ void proj_kernel(const _Float16* __restrict__ oh,
                            const _Float16* __restrict__ woT,
                            const float* __restrict__ bo,
                            float* __restrict__ out) {
  const int wave  = threadIdx.x >> 5;
  const int lane  = threadIdx.x & 31;
  const int row0  = blockIdx.x * 128 + wave * 32;
  const int nbase = blockIdx.y * 64;
  const int KD    = Hh * DVv;  // 1024

  v8f acc[2][4];
#pragma unroll
  for (int rb = 0; rb < 2; ++rb)
#pragma unroll
    for (int t = 0; t < 4; ++t) acc[rb][t] = v8f_zero();

  for (int kt = 0; kt < KD; kt += 32) {
    if (kt + 32 < KD) {
      __builtin_prefetch(oh + (size_t)(row0 + (lane & 15)) * KD + kt + 32, 0, 3);
      __builtin_prefetch(oh + (size_t)(row0 + 16 + (lane & 15)) * KD + kt + 32, 0, 3);
    }
    v16h a0 = load_a_frag(oh, KD, row0, kt);
    v16h a1 = load_a_frag(oh, KD, row0 + 16, kt);
    v16h bf[4];
#pragma unroll
    for (int t = 0; t < 4; ++t) bf[t] = load_b_frag(woT, KD, nbase + t * 16, kt);
#pragma unroll
    for (int t = 0; t < 4; ++t) {
      acc[0][t] = WMMA_F16(a0, bf[t], acc[0][t]);
      acc[1][t] = WMMA_F16(a1, bf[t], acc[1][t]);
    }
  }

  const int half = lane >> 4;
  const int nloc = lane & 15;
#pragma unroll
  for (int rb = 0; rb < 2; ++rb) {
#pragma unroll
    for (int t = 0; t < 4; ++t) {
      const int n = nbase + t * 16 + nloc;
      const float bb = bo[n];
#pragma unroll
      for (int r = 0; r < 8; ++r) {
        const int row = row0 + rb * 16 + r + 8 * half;
        out[(size_t)row * DMm + n] = acc[rb][t][r] + bb;
      }
    }
  }
}

// ---------------------------------------------------------------------------
extern "C" void kernel_launch(void* const* d_in, const int* in_sizes, int n_in,
                              void* d_out, int out_size, void* d_ws, size_t ws_size,
                              hipStream_t stream) {
  (void)in_sizes; (void)n_in; (void)out_size; (void)ws_size;
  const float*         x  = (const float*)d_in[0];
  const unsigned char* mk = (const unsigned char*)d_in[1];  // bool [B,S]
  const float* Wq = (const float*)d_in[2];
  const float* bq = (const float*)d_in[3];
  const float* Wk = (const float*)d_in[4];
  const float* bk = (const float*)d_in[5];
  const float* Wv = (const float*)d_in[6];
  const float* bv = (const float*)d_in[7];
  const float* Wo = (const float*)d_in[8];
  const float* bo = (const float*)d_in[9];

  float* out = (float*)d_out;                       // [B,S,DM]
  float* wt  = out + (size_t)Bb * Ss * DMm;         // [B,H,S,S]

  // Workspace carve-out (~50 MB of f16 intermediates; L2-resident).
  char* ws = (char*)d_ws;
  size_t off = 0;
  auto carve = [&](size_t nhalf) {
    _Float16* p = (_Float16*)(ws + off);
    off += nhalf * sizeof(_Float16);
    return p;
  };
  _Float16* xh  = carve((size_t)Bb * Ss * DMm);        // x in f16
  _Float16* wT  = carve((size_t)3 * Hh * DMm * DKk);   // Wq/Wk/Wv transposed [3][H][DK][DM]
  _Float16* woT = carve((size_t)DMm * DMm);            // Wo transposed [DM][H*DV]
  _Float16* qh  = carve((size_t)Bb * Hh * Ss * DKk);   // Q [B,H,S,DK]
  _Float16* kh  = carve((size_t)Bb * Hh * Ss * DKk);   // K [B,H,S,DK]
  _Float16* vT  = carve((size_t)Bb * Hh * DVv * Ss);   // V transposed [B,H,DV,S]
  _Float16* oh  = carve((size_t)Bb * Ss * Hh * DVv);   // concat heads [B,S,H*DV]

  convert_kernel<<<dim3(512), dim3(256), 0, stream>>>(x, Wq, Wk, Wv, Wo, xh, wT, woT);
  qkv_kernel<<<dim3((Bb * Ss) / 128, Hh, 3), dim3(128), 0, stream>>>(
      xh, wT, bq, bk, bv, qh, kh, vT);
  attn_kernel<<<dim3(Ss / 64, Hh, Bb), dim3(128), (size_t)64 * Ss * sizeof(_Float16), stream>>>(
      qh, kh, vT, mk, wt, oh);
  proj_kernel<<<dim3((Bb * Ss) / 128, DMm / 64), dim3(128), 0, stream>>>(oh, woT, bo, out);
}
